// OmegaSovereignReflex_15822659518960
// MI455X (gfx1250) — matrix operand order repaired and checked by reading, not compiled, not run on hardware
//
#include <hip/hip_runtime.h>
#include <hip/hip_bf16.h>

// ---------------------------------------------------------------------------
// Types for CDNA5 WMMA
// ---------------------------------------------------------------------------
typedef __attribute__((ext_vector_type(8)))  __bf16 v8bf_t;
typedef __attribute__((ext_vector_type(16))) __bf16 v16bf_t;
typedef __attribute__((ext_vector_type(8)))  float  v8f_t;

#define SEQ_LEN 1024
#define DMODEL  2048
#define DFFN    8192
#define NVOCAB  50257
#define NHEADS  32
#define HDIM    64

__device__ __forceinline__ unsigned short f32_to_bf16(float f) {
    unsigned u = __builtin_bit_cast(unsigned, f);
    u += 0x7FFFu + ((u >> 16) & 1u);          // round-to-nearest-even
    return (unsigned short)(u >> 16);
}

// ---------------------------------------------------------------------------
// Generic bf16 WMMA GEMM:  C[M,N] = A[M,K] * W[N,K]^T   (both row-major, f32)
// MODE 0: C = AB      MODE 1: C = AB + R[M,N]      MODE 2: C += R[row*4] * AB
// 128x128x32 tiles, 256 threads (8 waves, 2x4 wave grid).
// Software-pipelined: double-buffered LDS (ping-pong), next tile's global
// loads issued before current tile's WMMAs, tile t+2 prefetched.
// ---------------------------------------------------------------------------
template<int MODE>
__global__ __launch_bounds__(256)
void gemm_bf16_wmma(const float* __restrict__ A, const float* __restrict__ W,
                    float* __restrict__ C, const float* __restrict__ R,
                    int M, int N, int K)
{
    __shared__ __align__(16) unsigned short sA[2][128][40];  // 2 x 10 KB
    __shared__ __align__(16) unsigned short sB[2][128][40];

    const int tid  = threadIdx.x;
    const int wave = tid >> 5;
    const int lane = tid & 31;
    const int wm   = wave >> 2;      // 0..1  (64 rows each)
    const int wn   = wave & 3;       // 0..3  (32 cols each)
    const int grp  = lane >> 4;      // K-group select per ISA layout
    const int lr   = lane & 15;
    const int mBase = blockIdx.y * 128;
    const int nBase = blockIdx.x * 128;

    const v8f_t vz = {0.f,0.f,0.f,0.f,0.f,0.f,0.f,0.f};
    v8f_t acc[4][2];
#pragma unroll
    for (int mi = 0; mi < 4; ++mi)
#pragma unroll
        for (int ni = 0; ni < 2; ++ni) acc[mi][ni] = vz;

    const int ldRow = tid >> 1;        // 0..127
    const int ldCol = (tid & 1) * 16;  // 0 or 16
    const int gr = mBase + ldRow;      // global A row this thread stages
    const int gc = nBase + ldRow;      // global W row this thread stages
    const float* aRow = A + (size_t)gr * K + ldCol;
    const float* wRow = W + (size_t)gc * K + ldCol;

    float ra[16], rb[16];

    // ---- prologue: stage tile 0 ----
#pragma unroll
    for (int i = 0; i < 16; ++i) ra[i] = (gr < M) ? aRow[i] : 0.f;
#pragma unroll
    for (int i = 0; i < 16; ++i) rb[i] = (gc < N) ? wRow[i] : 0.f;
#pragma unroll
    for (int i = 0; i < 16; ++i) sA[0][ldRow][ldCol + i] = f32_to_bf16(ra[i]);
#pragma unroll
    for (int i = 0; i < 16; ++i) sB[0][ldRow][ldCol + i] = f32_to_bf16(rb[i]);
    __syncthreads();

    const int nk = K >> 5;
    for (int t = 0; t < nk; ++t) {
        const int buf = t & 1;

        // ---- issue next tile's global loads early (overlap with WMMA) ----
        if (t + 1 < nk) {
            const float* an = aRow + (size_t)(t + 1) * 32;
            const float* wn2 = wRow + (size_t)(t + 1) * 32;
#pragma unroll
            for (int i = 0; i < 16; ++i) ra[i] = (gr < M) ? an[i] : 0.f;
#pragma unroll
            for (int i = 0; i < 16; ++i) rb[i] = (gc < N) ? wn2[i] : 0.f;
            if (t + 2 < nk) {
                if (gr < M) __builtin_prefetch(aRow + (size_t)(t + 2) * 32, 0, 3);
                if (gc < N) __builtin_prefetch(wRow + (size_t)(t + 2) * 32, 0, 3);
            }
        }

        // ---- fragment loads (ISA 7.12.2 16-bit layouts) ----
        v16bf_t afr[4], bfr[2];
#pragma unroll
        for (int mi = 0; mi < 4; ++mi) {
            int row = wm * 64 + mi * 16 + lr;
            v8bf_t lo = *(const v8bf_t*)&sA[buf][row][grp * 8];       // K {0..7}/{8..15}
            v8bf_t hi = *(const v8bf_t*)&sA[buf][row][16 + grp * 8];  // K {16..23}/{24..31}
#pragma unroll
            for (int i = 0; i < 8; ++i) { afr[mi][i] = lo[i]; afr[mi][8 + i] = hi[i]; }
        }
#pragma unroll
        for (int ni = 0; ni < 2; ++ni) {
            int col = wn * 32 + ni * 16 + lr;
            v8bf_t lo = *(const v8bf_t*)&sB[buf][col][grp * 8];       // K {0..7}/{8..15}
            v8bf_t hi = *(const v8bf_t*)&sB[buf][col][16 + grp * 8];  // K {16..23}/{24..31}
#pragma unroll
            for (int i = 0; i < 8; ++i) { bfr[ni][i] = lo[i]; bfr[ni][8 + i] = hi[i]; }
        }

#pragma unroll
        for (int mi = 0; mi < 4; ++mi)
#pragma unroll
            for (int ni = 0; ni < 2; ++ni)
                acc[mi][ni] = __builtin_amdgcn_wmma_f32_16x16x32_bf16(
                    false, afr[mi], false, bfr[ni], (short)0, acc[mi][ni], false, false);

        // ---- convert + store next tile into the other buffer ----
        if (t + 1 < nk) {
            const int nb = buf ^ 1;
#pragma unroll
            for (int i = 0; i < 16; ++i) sA[nb][ldRow][ldCol + i] = f32_to_bf16(ra[i]);
#pragma unroll
            for (int i = 0; i < 16; ++i) sB[nb][ldRow][ldCol + i] = f32_to_bf16(rb[i]);
        }
        __syncthreads();   // one barrier per K-step (ping-pong safety + visibility)
    }

    // ---- epilogue (C/D layout: vgpr r -> M=r (lanes 0-15) / M=8+r; N = lane&15) ----
#pragma unroll
    for (int mi = 0; mi < 4; ++mi) {
#pragma unroll
        for (int ni = 0; ni < 2; ++ni) {
            int col = nBase + wn * 32 + ni * 16 + lr;
            if (col < N) {
#pragma unroll
                for (int rr = 0; rr < 8; ++rr) {
                    int row = mBase + wm * 64 + mi * 16 + grp * 8 + rr;
                    if (row < M) {
                        size_t idx = (size_t)row * N + col;
                        float v = acc[mi][ni][rr];
                        if (MODE == 0)      C[idx] = v;
                        else if (MODE == 1) C[idx] = v + R[idx];
                        else                C[idx] += R[(size_t)row * 4] * v;
                    }
                }
            }
        }
    }
}

// ---------------------------------------------------------------------------
// Attention: one block per (query-block of 64, head). fp32 VALU (~1.5% of FLOPs)
// Reproduces the reference's per-key-block max-subtract + O/(sum p + 1e-6).
// ---------------------------------------------------------------------------
__global__ __launch_bounds__(256)
void attn_kernel(const float* __restrict__ Q, const float* __restrict__ Km,
                 const float* __restrict__ V, float* __restrict__ O)
{
    const int qb = blockIdx.x, h = blockIdx.y;
    __shared__ float qs[64][66];
    __shared__ float kv[64][66];      // K block for scores, then V block
    __shared__ float sc[64][66];
    __shared__ float redm[64][4], reds[64][4];

    const int tid = threadIdx.x;
    const int r   = tid >> 2;         // row 0..63
    const int c4  = tid & 3;
    const int base = c4 * 16;
    const float LN1E4 = 9.210340371976184f;  // ln(10000)

    // load q block with RoPE
    {
        int pos = qb * 64 + r;
        size_t g = (size_t)pos * DMODEL + h * HDIM + base;
#pragma unroll
        for (int dd = 0; dd < 16; dd += 2) {
            float x0 = Q[g + dd], x1 = Q[g + dd + 1];
            int p = (base + dd) >> 1;
            float inv = __expf(-(float)p * (LN1E4 / 32.0f));
            float ang = (float)pos * inv;
            float cc = cosf(ang), ss = sinf(ang);
            qs[r][base + dd]     = x0 * cc - x1 * ss;
            qs[r][base + dd + 1] = x0 * ss + x1 * cc;
        }
    }

    float oacc[16];
#pragma unroll
    for (int i = 0; i < 16; ++i) oacc[i] = 0.f;
    float Lacc = 0.f;

    for (int jb = 0; jb <= qb; ++jb) {
        // load K block with RoPE
        {
            int pos = jb * 64 + r;
            size_t g = (size_t)pos * DMODEL + h * HDIM + base;
#pragma unroll
            for (int dd = 0; dd < 16; dd += 2) {
                float x0 = Km[g + dd], x1 = Km[g + dd + 1];
                int p = (base + dd) >> 1;
                float inv = __expf(-(float)p * (LN1E4 / 32.0f));
                float ang = (float)pos * inv;
                float cc = cosf(ang), ss = sinf(ang);
                kv[r][base + dd]     = x0 * cc - x1 * ss;
                kv[r][base + dd + 1] = x0 * ss + x1 * cc;
            }
        }
        __syncthreads();

        // scores for row r, cols base..base+15
        float pv[16];
        float mloc = -3.4e38f;
#pragma unroll 4
        for (int c = 0; c < 16; ++c) {
            int col = base + c;
            float d = 0.f;
            for (int k = 0; k < 64; ++k) d += qs[r][k] * kv[col][k];
            d *= 0.125f;                              // hd^-0.5
            if (jb == qb && col > r) d = -3.4e38f;    // causal in diag block
            pv[c] = d;
            if (d > mloc) mloc = d;
        }
        redm[r][c4] = mloc;
        __syncthreads();
        float m = fmaxf(fmaxf(redm[r][0], redm[r][1]), fmaxf(redm[r][2], redm[r][3]));

        float sloc = 0.f;
#pragma unroll
        for (int c = 0; c < 16; ++c) {
            float p = (pv[c] <= -3.3e38f) ? 0.f : __expf(pv[c] - m);
            sc[r][base + c] = p;
            sloc += p;
        }
        reds[r][c4] = sloc;
        __syncthreads();
        Lacc += reds[r][0] + reds[r][1] + reds[r][2] + reds[r][3];

        // stage V block (overwrites K block)
        {
            size_t g = (size_t)(jb * 64 + r) * DMODEL + h * HDIM + base;
#pragma unroll
            for (int dd = 0; dd < 16; ++dd) kv[r][base + dd] = V[g + dd];
        }
        __syncthreads();

        // O[r][base..base+15] += p . V
        for (int k = 0; k < 64; ++k) {
            float p = sc[r][k];
#pragma unroll
            for (int dd = 0; dd < 16; ++dd) oacc[dd] += p * kv[k][base + dd];
        }
        __syncthreads();
    }

    float invL = 1.0f / (Lacc + 1e-6f);
    size_t g = (size_t)(qb * 64 + r) * DMODEL + h * HDIM + base;
#pragma unroll
    for (int dd = 0; dd < 16; ++dd) O[g + dd] = oacc[dd] * invL;
}

// ---------------------------------------------------------------------------
// Small kernels
// ---------------------------------------------------------------------------
__global__ void embed_kernel(const int* __restrict__ tok, const float* __restrict__ emb,
                             float* __restrict__ X)
{
    int s = blockIdx.x;
    int t = tok[s];
    for (int d = threadIdx.x; d < DMODEL; d += blockDim.x)
        X[(size_t)s * DMODEL + d] = emb[(size_t)t * DMODEL + d];
}

__global__ void ln_kernel(float* __restrict__ x, const float* __restrict__ g,
                          const float* __restrict__ b)
{
    __shared__ float row[DMODEL];
    __shared__ float red[256];
    const int t = threadIdx.x;
    size_t base = (size_t)blockIdx.x * DMODEL;

    float s = 0.f;
    for (int i = t; i < DMODEL; i += 256) { float v = x[base + i]; row[i] = v; s += v; }
    red[t] = s; __syncthreads();
    for (int off = 128; off > 0; off >>= 1) { if (t < off) red[t] += red[t + off]; __syncthreads(); }
    float mu = red[0] / (float)DMODEL;
    __syncthreads();

    float s2 = 0.f;
    for (int i = t; i < DMODEL; i += 256) { float d = row[i] - mu; s2 += d * d; }
    red[t] = s2; __syncthreads();
    for (int off = 128; off > 0; off >>= 1) { if (t < off) red[t] += red[t + off]; __syncthreads(); }
    float inv = rsqrtf(red[0] / (float)DMODEL + 1e-5f);
    __syncthreads();

    for (int i = t; i < DMODEL; i += 256)
        x[base + i] = (row[i] - mu) * inv * g[i] + b[i];
}

__global__ void gate_kernel(const float* __restrict__ x, const float* __restrict__ gw,
                            float* __restrict__ wdense, float* __restrict__ vars)
{
    __shared__ float red[4][256];
    const int t = threadIdx.x;
    size_t base = (size_t)blockIdx.x * DMODEL;
    float a0 = 0.f, a1 = 0.f, a2 = 0.f, a3 = 0.f;
    for (int i = t; i < DMODEL; i += 256) {
        float v = x[base + i];
        a0 += v * gw[i];
        a1 += v * gw[DMODEL + i];
        a2 += v * gw[2 * DMODEL + i];
        a3 += v * gw[3 * DMODEL + i];
    }
    red[0][t] = a0; red[1][t] = a1; red[2][t] = a2; red[3][t] = a3;
    __syncthreads();
    for (int off = 128; off > 0; off >>= 1) {
        if (t < off)
            for (int e = 0; e < 4; ++e) red[e][t] += red[e][t + off];
        __syncthreads();
    }
    if (t == 0) {
        float l[4] = { red[0][0], red[1][0], red[2][0], red[3][0] };
        float mu = 0.25f * (l[0] + l[1] + l[2] + l[3]);
        float v = 0.f;
        for (int e = 0; e < 4; ++e) { float d = l[e] - mu; v += d * d; }
        vars[blockIdx.x] = v / 3.0f;                      // ddof=1
        float m = fmaxf(fmaxf(l[0], l[1]), fmaxf(l[2], l[3]));
        float p[4], ps = 0.f;
        for (int e = 0; e < 4; ++e) { p[e] = __expf(l[e] - m); ps += p[e]; }
        for (int e = 0; e < 4; ++e) p[e] /= ps;
        int e1 = 0;
        for (int e = 1; e < 4; ++e) if (p[e] > p[e1]) e1 = e;    // first on ties
        int e2 = -1;
        for (int e = 0; e < 4; ++e) if (e != e1 && (e2 < 0 || p[e] > p[e2])) e2 = e;
        float ws = p[e1] + p[e2];
        float outw[4] = {0.f, 0.f, 0.f, 0.f};
        outw[e1] = p[e1] / ws; outw[e2] = p[e2] / ws;
        for (int e = 0; e < 4; ++e) wdense[(size_t)blockIdx.x * 4 + e] = outw[e];
    }
}

__global__ void zero_kernel(float* p, long n) {
    long i = (long)blockIdx.x * blockDim.x + threadIdx.x;
    long st = (long)gridDim.x * blockDim.x;
    for (; i < n; i += st) p[i] = 0.f;
}

__global__ void add_kernel(float* __restrict__ x, const float* __restrict__ y, long n) {
    long i = (long)blockIdx.x * blockDim.x + threadIdx.x;
    long st = (long)gridDim.x * blockDim.x;
    for (; i < n; i += st) x[i] += y[i];
}

__global__ void silu_mul_kernel(float* __restrict__ h1, const float* __restrict__ h3, long n) {
    long i = (long)blockIdx.x * blockDim.x + threadIdx.x;
    long st = (long)gridDim.x * blockDim.x;
    for (; i < n; i += st) {
        float a = h1[i];
        h1[i] = (a / (1.0f + __expf(-a))) * h3[i];
    }
}

__global__ void aux_kernel(const float* __restrict__ vars, float* __restrict__ out) {
    __shared__ float red[256];
    const int t = threadIdx.x;
    float s = 0.f;
    for (int i = t; i < SEQ_LEN; i += 256) s += vars[i];
    red[t] = s; __syncthreads();
    for (int off = 128; off > 0; off >>= 1) { if (t < off) red[t] += red[t + off]; __syncthreads(); }
    if (t == 0) out[0] = red[0] / (float)SEQ_LEN;
}

// ---------------------------------------------------------------------------
// Host-side orchestration
// ---------------------------------------------------------------------------
static inline void launch_gemm(int mode, const float* A, const float* W, float* C,
                               const float* R, int M, int N, int K, hipStream_t st)
{
    dim3 g((N + 127) / 128, (M + 127) / 128);
    if (mode == 0)      gemm_bf16_wmma<0><<<g, 256, 0, st>>>(A, W, C, R, M, N, K);
    else if (mode == 1) gemm_bf16_wmma<1><<<g, 256, 0, st>>>(A, W, C, R, M, N, K);
    else                gemm_bf16_wmma<2><<<g, 256, 0, st>>>(A, W, C, R, M, N, K);
}

extern "C" void kernel_launch(void* const* d_in, const int* in_sizes, int n_in,
                              void* d_out, int out_size, void* d_ws, size_t ws_size,
                              hipStream_t stream)
{
    (void)in_sizes; (void)n_in; (void)out_size; (void)ws_size;
    const int*   tokens = (const int*)  d_in[0];
    const float* emb    = (const float*)d_in[1];
    const float* wq     = (const float*)d_in[2];
    const float* wk     = (const float*)d_in[3];
    const float* wv     = (const float*)d_in[4];
    const float* wo     = (const float*)d_in[5];
    const float* ln1_g  = (const float*)d_in[6];
    const float* ln1_b  = (const float*)d_in[7];
    const float* gate_w = (const float*)d_in[8];
    const float* w1     = (const float*)d_in[9];
    const float* w2     = (const float*)d_in[10];
    const float* w3     = (const float*)d_in[11];
    const float* ln2_g  = (const float*)d_in[12];
    const float* ln2_b  = (const float*)d_in[13];
    const float* fin_g  = (const float*)d_in[14];
    const float* fin_b  = (const float*)d_in[15];
    const float* head_w = (const float*)d_in[16];
    float* out = (float*)d_out;
    float* ws  = (float*)d_ws;

    const int S = SEQ_LEN, D = DMODEL, F = DFFN, Vc = NVOCAB;
    size_t o = 0;
    float* X  = ws + o; o += (size_t)S * D;
    float* Qb = ws + o; o += (size_t)S * D;
    float* Kb = ws + o; o += (size_t)S * D;
    float* Vb = ws + o; o += (size_t)S * D;
    float* AO = ws + o; o += (size_t)S * D;
    float* H1 = ws + o; o += (size_t)S * F;
    float* H3 = ws + o; o += (size_t)S * F;
    float* MO = ws + o; o += (size_t)S * D;
    float* GW = ws + o; o += (size_t)S * 4;
    float* VR = ws + o; o += (size_t)S;

    // 1) embedding
    embed_kernel<<<S, 256, 0, stream>>>(tokens, emb, X);

    // 2) Q,K,V projections (bf16 WMMA)
    launch_gemm(0, X, wq, Qb, nullptr, S, D, D, stream);
    launch_gemm(0, X, wk, Kb, nullptr, S, D, D, stream);
    launch_gemm(0, X, wv, Vb, nullptr, S, D, D, stream);

    // 3) attention (RoPE inside)
    attn_kernel<<<dim3(S / 64, NHEADS), 256, 0, stream>>>(Qb, Kb, Vb, AO);

    // 4) output proj + residual:  X = AO @ wo^T + X
    launch_gemm(1, AO, wo, X, X, S, D, D, stream);

    // 5) LN1 (in-place)
    ln_kernel<<<S, 256, 0, stream>>>(X, ln1_g, ln1_b);

    // 6) gating: softmax/top-2 weights + per-token logit variance
    gate_kernel<<<S, 256, 0, stream>>>(X, gate_w, GW, VR);

    // 7) MoE: dense over all experts, weighted accumulate into MO
    zero_kernel<<<1024, 256, 0, stream>>>(MO, (long)S * D);
    for (int e = 0; e < 4; ++e) {
        const float* w1e = w1 + (size_t)e * F * D;
        const float* w3e = w3 + (size_t)e * F * D;
        const float* w2e = w2 + (size_t)e * D * F;
        launch_gemm(0, X, w1e, H1, nullptr, S, F, D, stream);
        launch_gemm(0, X, w3e, H3, nullptr, S, F, D, stream);
        silu_mul_kernel<<<4096, 256, 0, stream>>>(H1, H3, (long)S * F);
        launch_gemm(2, H1, w2e, MO, GW + e, S, D, F, stream);  // MO += w[t,e] * (H @ w2^T)
    }

    // 8) residual + LN2 + final LN
    add_kernel<<<1024, 256, 0, stream>>>(X, MO, (long)S * D);
    ln_kernel<<<S, 256, 0, stream>>>(X, ln2_g, ln2_b);
    ln_kernel<<<S, 256, 0, stream>>>(X, fin_g, fin_b);

    // 9) vocab head -> logits (N=50257, bounds-checked tiles)
    launch_gemm(0, X, head_w, out, nullptr, S, Vc, D, stream);

    // 10) aux = mean(var(gate logits, ddof=1))
    aux_kernel<<<1, 256, 0, stream>>>(VR, out + (size_t)S * Vc);
}
